// SharpGentleBlock_81870666596742
// MI455X (gfx1250) — matrix operand order repaired and checked by reading, not compiled, not run on hardware
//
#include <hip/hip_runtime.h>
#include <math.h>

typedef __bf16 bf16;
typedef __attribute__((ext_vector_type(16))) __bf16 v16bf;
typedef __attribute__((ext_vector_type(8)))  float   v8f;

constexpr int  Bsz  = 16;
constexpr int  Cch  = 256;
constexpr int  Npt  = 4096;
constexpr int  Msel = 1024;
constexpr float EPSBN = 1e-5f;

// ---------------------------------------------------------------------------
// Fragment loaders per CDNA5 ISA 7.12.2 (bf16 16x16x32):
//   A lane: row = lane&15, e0..7 @ k+8g, e8..15 @ k+16+8g   (g = lane>>4)
//   B lane: col = lane&15, e0..7 @ k+8g, e8..15 @ k+16+8g   (g = lane>>4)
// ---------------------------------------------------------------------------
__device__ __forceinline__ v16bf load_frag_a(const bf16* __restrict__ base,
                                             int ld, int k, int rc, int g) {
    union { v16bf v; float4 f[2]; } u;
    const bf16* p = base + (long)rc * ld + k + 8 * g;
    u.f[0] = *reinterpret_cast<const float4*>(p);
    u.f[1] = *reinterpret_cast<const float4*>(p + 16);
    return u.v;
}
__device__ __forceinline__ v16bf load_frag_b(const bf16* __restrict__ base,
                                             int ld, int k, int rc, int g) {
    union { v16bf v; float4 f[2]; } u;
    const bf16* p = base + (long)rc * ld + k + 8 * g;
    u.f[0] = *reinterpret_cast<const float4*>(p);
    u.f[1] = *reinterpret_cast<const float4*>(p + 16);
    return u.v;
}

// ---------------------------------------------------------------------------
// Generic bf16 GEMM:  D[i][j] = alpha * sum_k A[i][k]*B[j][k]  (+bias)(+resid)
// Register-blocked: each wave computes a 32x64 tile (2 i-tiles x 4 j-tiles,
// 8 v_wmma per k-step, 2 A-frags + 4 B-frags -> ~21 flops/byte of frag
// traffic vs 8 for a single-tile wave).
// Block = 256 threads = 8 waves arranged 2(i) x 4(j) -> 64x256 macro-tile.
// ---------------------------------------------------------------------------
__global__ __launch_bounds__(256)
void gemm_bf16_kernel(const bf16* __restrict__ A, int lda, long sA,
                      const bf16* __restrict__ Bm, int ldb, long sB,
                      void* __restrict__ Dst, int ldd, long sD,
                      const float* __restrict__ bias, int bias_mode, // 0 none,1 over j,2 over i
                      const bf16* __restrict__ resid, int ldr, long sR,
                      float alpha, int out_f32, int Kk)
{
    const int lane = threadIdx.x & 31;
    const int wave = threadIdx.x >> 5;
    const int wi = wave >> 2, wj = wave & 3;
    const int i0 = blockIdx.y * 64 + wi * 32;     // wave covers rows i0..i0+31
    const int j0 = blockIdx.x * 256 + wj * 64;    // wave covers cols j0..j0+63
    const int bz = blockIdx.z;

    const int rc = lane & 15;
    const int g  = lane >> 4;

    const bf16* A0 = A  + (long)bz * sA + (long)i0 * lda;
    const bf16* A1 = A0 + (long)16 * lda;
    const bf16* Bb = Bm + (long)bz * sB + (long)j0 * ldb;

    v8f acc[2][4] = {};
    for (int k = 0; k < Kk; k += 32) {
        __builtin_prefetch(A0 + (long)rc * lda + k + 32, 0, 1);
        v16bf a0 = load_frag_a(A0, lda, k, rc, g);
        v16bf a1 = load_frag_a(A1, lda, k, rc, g);
        v16bf b0 = load_frag_b(Bb,                 ldb, k, rc, g);
        v16bf b1 = load_frag_b(Bb + (long)16 * ldb, ldb, k, rc, g);
        v16bf b2 = load_frag_b(Bb + (long)32 * ldb, ldb, k, rc, g);
        v16bf b3 = load_frag_b(Bb + (long)48 * ldb, ldb, k, rc, g);
        acc[0][0] = __builtin_amdgcn_wmma_f32_16x16x32_bf16(false, a0, false, b0, (short)0, acc[0][0], false, false);
        acc[0][1] = __builtin_amdgcn_wmma_f32_16x16x32_bf16(false, a0, false, b1, (short)0, acc[0][1], false, false);
        acc[0][2] = __builtin_amdgcn_wmma_f32_16x16x32_bf16(false, a0, false, b2, (short)0, acc[0][2], false, false);
        acc[0][3] = __builtin_amdgcn_wmma_f32_16x16x32_bf16(false, a0, false, b3, (short)0, acc[0][3], false, false);
        acc[1][0] = __builtin_amdgcn_wmma_f32_16x16x32_bf16(false, a1, false, b0, (short)0, acc[1][0], false, false);
        acc[1][1] = __builtin_amdgcn_wmma_f32_16x16x32_bf16(false, a1, false, b1, (short)0, acc[1][1], false, false);
        acc[1][2] = __builtin_amdgcn_wmma_f32_16x16x32_bf16(false, a1, false, b2, (short)0, acc[1][2], false, false);
        acc[1][3] = __builtin_amdgcn_wmma_f32_16x16x32_bf16(false, a1, false, b3, (short)0, acc[1][3], false, false);
    }

#pragma unroll
    for (int ii = 0; ii < 2; ++ii) {
#pragma unroll
        for (int jj = 0; jj < 4; ++jj) {
#pragma unroll
            for (int r = 0; r < 8; ++r) {
                const int i = i0 + ii * 16 + r + 8 * g;
                const int j = j0 + jj * 16 + rc;
                float v = acc[ii][jj][r] * alpha;
                if (bias_mode == 1)      v += bias[j];
                else if (bias_mode == 2) v += bias[i];
                if (resid) v += (float)resid[(long)bz * sR + (long)i * ldr + j];
                if (out_f32) ((float*)Dst)[(long)bz * sD + (long)i * ldd + j] = v;
                else         ((bf16*) Dst)[(long)bz * sD + (long)i * ldd + j] = (bf16)v;
            }
        }
    }
}

// --------------------------- helper kernels --------------------------------

__global__ __launch_bounds__(256)
void cvt_bf16_kernel(bf16* dst, const float* src, int n) {
    int i = blockIdx.x * 256 + threadIdx.x;
    if (i < n) dst[i] = (bf16)src[i];
}

// x [B][C][N] fp32  ->  xT [B][N][C] bf16  (32x32 LDS tile transpose)
__global__ __launch_bounds__(256)
void transpose_kernel(const float* __restrict__ x, bf16* __restrict__ xT) {
    __shared__ float tile[32][33];
    const int b  = blockIdx.z;
    const int n0 = blockIdx.x * 32;
    const int c0 = blockIdx.y * 32;
    for (int r = threadIdx.y; r < 32; r += 8)
        tile[r][threadIdx.x] = x[((long)b * Cch + c0 + r) * Npt + n0 + threadIdx.x];
    __syncthreads();
    for (int r = threadIdx.y; r < 32; r += 8)
        xT[((long)b * Npt + n0 + r) * Cch + c0 + threadIdx.x] = (bf16)tile[threadIdx.x][r];
}

// mean over N for each (b,c)
__global__ __launch_bounds__(256)
void mean_kernel(const float* __restrict__ x, float* __restrict__ mu) {
    const long row = blockIdx.x;           // b*C + c
    const float* p = x + row * (long)Npt;
    __shared__ float red[256];
    float s = 0.f;
    for (int n = threadIdx.x; n < Npt; n += 256) s += p[n];
    red[threadIdx.x] = s; __syncthreads();
    for (int st = 128; st > 0; st >>= 1) {
        if (threadIdx.x < st) red[threadIdx.x] += red[threadIdx.x + st];
        __syncthreads();
    }
    if (threadIdx.x == 0) mu[row] = red[0] * (1.0f / Npt);
}

// score[b][n] = || x[b][:,n] - mu[b][:] ||_2
__global__ __launch_bounds__(256)
void score_kernel(const float* __restrict__ x, const float* __restrict__ mu,
                  float* __restrict__ score) {
    const int n = blockIdx.x * 256 + threadIdx.x;
    const int b = blockIdx.y;
    const float* xb = x + (long)b * Cch * Npt;
    const float* mb = mu + b * Cch;
    float acc = 0.f;
    for (int c = 0; c < Cch; ++c) {
        float d = xb[(long)c * Npt + n] - mb[c];
        acc += d * d;
    }
    score[(long)b * Npt + n] = sqrtf(acc);
}

// exact top-M / bottom-M selection by rank counting (tie-break: lower index)
__global__ __launch_bounds__(256)
void select_kernel(const float* __restrict__ score,
                   int* __restrict__ idx_s, int* __restrict__ idx_g) {
    const int b = blockIdx.x;
    __shared__ float sc[Npt];
    __shared__ int cnt_s, cnt_g;
    for (int n = threadIdx.x; n < Npt; n += 256) sc[n] = score[(long)b * Npt + n];
    if (threadIdx.x == 0) { cnt_s = 0; cnt_g = 0; }
    __syncthreads();
    for (int n = threadIdx.x; n < Npt; n += 256) {
        const float s = sc[n];
        int hi = 0, lo = 0, eq = 0;
        for (int j = 0; j < Npt; ++j) {
            const float t = sc[j];
            hi += (t > s);
            lo += (t < s);
            eq += ((t == s) && (j < n));
        }
        if (hi + eq < Msel) { int p = atomicAdd(&cnt_s, 1); idx_s[b * Msel + p] = n; }
        if (lo + eq < Msel) { int p = atomicAdd(&cnt_g, 1); idx_g[b * Msel + p] = n; }
    }
}

// ctxT[b][m][c] = bf16( x[b][c][idx[b][m]] )
__global__ __launch_bounds__(256)
void gather_kernel(const float* __restrict__ x, const int* __restrict__ idx,
                   bf16* __restrict__ ctxT) {
    const int c = threadIdx.x;
    const int m = blockIdx.x;
    const int b = blockIdx.y;
    const int n = idx[b * Msel + m];
    ctxT[((long)b * Msel + m) * Cch + c] = (bf16)x[((long)b * Cch + c) * Npt + n];
}

// row softmax over M, bf16 in/out, one block per (b,n) row
__global__ __launch_bounds__(256)
void softmax_kernel(bf16* __restrict__ S, int Mc) {
    const int tid = threadIdx.x;
    bf16* p = S + (long)blockIdx.x * Mc;
    __shared__ float red[256];
    float mx = -3.4e38f;
    for (int j = tid; j < Mc; j += 256) mx = fmaxf(mx, (float)p[j]);
    red[tid] = mx; __syncthreads();
    for (int s = 128; s > 0; s >>= 1) {
        if (tid < s) red[tid] = fmaxf(red[tid], red[tid + s]);
        __syncthreads();
    }
    mx = red[0]; __syncthreads();
    float sum = 0.f;
    for (int j = tid; j < Mc; j += 256) {
        float e = __expf((float)p[j] - mx);
        p[j] = (bf16)e;
        sum += e;
    }
    red[tid] = sum; __syncthreads();
    for (int s = 128; s > 0; s >>= 1) {
        if (tid < s) red[tid] += red[tid + s];
        __syncthreads();
    }
    const float inv = 1.0f / red[0];
    for (int j = tid; j < Mc; j += 256) p[j] = (bf16)((float)p[j] * inv);
}

// per-channel batch stats over (B,N) of F [B][N][C]
__global__ __launch_bounds__(256)
void bn_stats_kernel(const float* __restrict__ F, float* __restrict__ mu,
                     float* __restrict__ rs) {
    const int d = blockIdx.x;
    __shared__ float r1[256], r2[256];
    float s = 0.f, s2 = 0.f;
    for (long r = threadIdx.x; r < (long)Bsz * Npt; r += 256) {
        float v = F[r * Cch + d];
        s += v; s2 += v * v;
    }
    r1[threadIdx.x] = s; r2[threadIdx.x] = s2; __syncthreads();
    for (int st = 128; st > 0; st >>= 1) {
        if (threadIdx.x < st) { r1[threadIdx.x] += r1[threadIdx.x + st];
                                r2[threadIdx.x] += r2[threadIdx.x + st]; }
        __syncthreads();
    }
    if (threadIdx.x == 0) {
        const float m = r1[0] / ((float)Bsz * Npt);
        const float var = r2[0] / ((float)Bsz * Npt) - m * m;
        mu[d] = m;
        rs[d] = rsqrtf(var + EPSBN);
    }
}

// BN apply + ReLU + transpose to output [B][C][N]
__global__ __launch_bounds__(256)
void bn_final_kernel(const float* __restrict__ F, const float* __restrict__ mu,
                     const float* __restrict__ rs, const float* __restrict__ gamma,
                     const float* __restrict__ beta, float* __restrict__ out) {
    const int n = blockIdx.x * 256 + threadIdx.x;
    const int d = blockIdx.y;
    const int b = blockIdx.z;
    float v = F[((long)b * Npt + n) * Cch + d];
    v = (v - mu[d]) * rs[d] * gamma[d] + beta[d];
    out[((long)b * Cch + d) * Npt + n] = fmaxf(v, 0.0f);
}

// ---------------------------------------------------------------------------
extern "C" void kernel_launch(void* const* d_in, const int* in_sizes, int n_in,
                              void* d_out, int out_size, void* d_ws, size_t ws_size,
                              hipStream_t stream)
{
    (void)in_sizes; (void)n_in; (void)out_size; (void)ws_size;
    const float* x = (const float*)d_in[0];
    // inputs 2..17: per-branch attention params, 18..21: fuse/BN
    const float* Wq[2] = { (const float*)d_in[2],  (const float*)d_in[10] };
    const float* bq[2] = { (const float*)d_in[3],  (const float*)d_in[11] };
    const float* Wk[2] = { (const float*)d_in[4],  (const float*)d_in[12] };
    const float* bk[2] = { (const float*)d_in[5],  (const float*)d_in[13] };
    const float* Wv[2] = { (const float*)d_in[6],  (const float*)d_in[14] };
    const float* bv[2] = { (const float*)d_in[7],  (const float*)d_in[15] };
    const float* Wo[2] = { (const float*)d_in[8],  (const float*)d_in[16] };
    const float* bo[2] = { (const float*)d_in[9],  (const float*)d_in[17] };
    const float* Wf    = (const float*)d_in[18];
    const float* bfb   = (const float*)d_in[19];
    const float* gamma = (const float*)d_in[20];
    const float* beta  = (const float*)d_in[21];

    char* ws = (char*)d_ws;
    size_t off = 0;
    auto alloc = [&](size_t bytes) -> char* {
        char* p = ws + off;
        off += (bytes + 255) & ~(size_t)255;
        return p;
    };

    float* muX   = (float*)alloc((size_t)Bsz * Cch * 4);
    float* score = (float*)alloc((size_t)Bsz * Npt * 4);
    int*   idx_s = (int*)  alloc((size_t)Bsz * Msel * 4);
    int*   idx_g = (int*)  alloc((size_t)Bsz * Msel * 4);
    bf16*  xT    = (bf16*) alloc((size_t)Bsz * Npt * Cch * 2);
    bf16*  Wqh[2], *Wkh[2], *Wvh[2], *Woh[2];
    for (int b = 0; b < 2; ++b) {
        Wqh[b] = (bf16*)alloc((size_t)Cch * Cch * 2);
        Wkh[b] = (bf16*)alloc((size_t)Cch * Cch * 2);
        Wvh[b] = (bf16*)alloc((size_t)Cch * Cch * 2);
        Woh[b] = (bf16*)alloc((size_t)Cch * Cch * 2);
    }
    bf16* Wfh  = (bf16*)alloc((size_t)Cch * 2 * Cch * 2);
    bf16* ctxT = (bf16*)alloc((size_t)Bsz * Msel * Cch * 2);
    bf16* Qb   = (bf16*)alloc((size_t)Bsz * Npt * Cch * 2);
    bf16* Kp   = (bf16*)alloc((size_t)Bsz * Msel * Cch * 2);
    bf16* VpT  = (bf16*)alloc((size_t)Bsz * Cch * Msel * 2);
    bf16* Sb   = (bf16*)alloc((size_t)Bsz * Npt * Msel * 2);
    bf16* Ob   = (bf16*)alloc((size_t)Bsz * Npt * Cch * 2);
    bf16* hT   = (bf16*)alloc((size_t)Bsz * Npt * 2 * Cch * 2);
    float* Fb  = (float*)alloc((size_t)Bsz * Npt * Cch * 4);
    float* bmu = (float*)alloc((size_t)Cch * 4);
    float* brs = (float*)alloc((size_t)Cch * 4);

    auto cvt = [&](bf16* dst, const float* src, int n) {
        cvt_bf16_kernel<<<dim3((n + 255) / 256), 256, 0, stream>>>(dst, src, n);
    };
    for (int b = 0; b < 2; ++b) {
        cvt(Wqh[b], Wq[b], Cch * Cch);
        cvt(Wkh[b], Wk[b], Cch * Cch);
        cvt(Wvh[b], Wv[b], Cch * Cch);
        cvt(Woh[b], Wo[b], Cch * Cch);
    }
    cvt(Wfh, Wf, Cch * 2 * Cch);

    transpose_kernel<<<dim3(Npt / 32, Cch / 32, Bsz), dim3(32, 8), 0, stream>>>(x, xT);
    mean_kernel<<<Bsz * Cch, 256, 0, stream>>>(x, muX);
    score_kernel<<<dim3(Npt / 256, Bsz), 256, 0, stream>>>(x, muX, score);
    select_kernel<<<Bsz, 256, 0, stream>>>(score, idx_s, idx_g);

    auto gemm = [&](const bf16* A, int lda, long sA,
                    const bf16* Bm, int ldb, long sB,
                    void* Dst, int ldd, long sD,
                    const float* bias, int bias_mode,
                    const bf16* resid, int ldr, long sR,
                    float alpha, int out_f32, int Mi, int Nj, int Kk) {
        dim3 grid(Nj / 256, Mi / 64, Bsz);
        gemm_bf16_kernel<<<grid, 256, 0, stream>>>(A, lda, sA, Bm, ldb, sB,
            Dst, ldd, sD, bias, bias_mode, resid, ldr, sR, alpha, out_f32, Kk);
    };

    const long sX  = (long)Npt * Cch;     // xT / Q / O batch stride
    const long sCx = (long)Msel * Cch;    // ctxT / Kp
    const long sV  = (long)Cch * Msel;    // VpT
    const long sS  = (long)Npt * Msel;    // scores / probs
    const long sH  = (long)Npt * 2 * Cch; // hT

    for (int br = 0; br < 2; ++br) {
        const int* idx = br ? idx_g : idx_s;
        gather_kernel<<<dim3(Msel, Bsz), Cch, 0, stream>>>(x, idx, ctxT);
        // Q[n][d] = xT · Wq^T + bq
        gemm(xT, Cch, sX, Wqh[br], Cch, 0, Qb, Cch, sX,
             bq[br], 1, nullptr, 0, 0, 1.0f, 0, Npt, Cch, Cch);
        // K[m][d] = ctxT · Wk^T + bk
        gemm(ctxT, Cch, sCx, Wkh[br], Cch, 0, Kp, Cch, sCx,
             bk[br], 1, nullptr, 0, 0, 1.0f, 0, Msel, Cch, Cch);
        // V^T[d][m] = Wv · ctxT^T + bv (bias over i=d)
        gemm(Wvh[br], Cch, 0, ctxT, Cch, sCx, VpT, Msel, sV,
             bv[br], 2, nullptr, 0, 0, 1.0f, 0, Cch, Msel, Cch);
        // S[n][m] = (Q·K^T)/sqrt(C)
        gemm(Qb, Cch, sX, Kp, Cch, sCx, Sb, Msel, sS,
             nullptr, 0, nullptr, 0, 0, 0.0625f, 0, Npt, Msel, Cch);
        softmax_kernel<<<Bsz * Npt, 256, 0, stream>>>(Sb, Msel);
        // O[n][d] = P · V
        gemm(Sb, Msel, sS, VpT, Msel, sV, Ob, Cch, sX,
             nullptr, 0, nullptr, 0, 0, 1.0f, 0, Npt, Cch, Msel);
        // hT[n][br*C + e] = O · Wo^T + bo + xT   (residual)
        gemm(Ob, Cch, sX, Woh[br], Cch, 0, hT + br * Cch, 2 * Cch, sH,
             bo[br], 1, xT, Cch, sX, 1.0f, 0, Npt, Cch, Cch);
    }

    // F[n][d] = hT · Wf^T + bf   (fp32 out for BN)
    gemm(hT, 2 * Cch, sH, Wfh, 2 * Cch, 0, Fb, Cch, sX,
         bfb, 1, nullptr, 0, 0, 1.0f, 1, Npt, Cch, 2 * Cch);
    bn_stats_kernel<<<Cch, 256, 0, stream>>>(Fb, bmu, brs);
    bn_final_kernel<<<dim3(Npt / 256, Cch, Bsz), 256, 0, stream>>>(
        Fb, bmu, brs, gamma, beta, (float*)d_out);
}
